// TPAAttention_86792699118322
// MI455X (gfx1250) — matrix operand order repaired and checked by reading, not compiled, not run on hardware
//
#include <hip/hip_runtime.h>

// ---------------------------------------------------------------------------
// Problem constants (from reference)
// ---------------------------------------------------------------------------
#define Bc_   2
#define Sc_   2048
#define HIDc_ 2048
#define Hc_   16
#define HKVc_ 8
#define Dc_   128
#define RQc_  6
#define RKc_  2
#define RVc_  2
#define WINDOWc_ 1024
#define SOFTCAPc_ 50.0f
#define SCALINGc_ 0.08838834764831845f   // 1/sqrt(128)

typedef __bf16 bf16;
typedef __attribute__((ext_vector_type(16))) __bf16 v16bf;
typedef __attribute__((ext_vector_type(8)))  __bf16 v8bf;
typedef __attribute__((ext_vector_type(8)))  float  v8f;

__device__ __forceinline__ v8f zero_v8f() {
    v8f v;
#pragma unroll
    for (int i = 0; i < 8; ++i) v[i] = 0.0f;
    return v;
}
__device__ __forceinline__ v8bf zero_v8bf() {
    v8bf v;
#pragma unroll
    for (int i = 0; i < 8; ++i) v[i] = (bf16)0.0f;
    return v;
}

// A-fragment (16x32 bf16, MxK): lane holds row m = lane%16.
// elements 0..7  <-> K = 8*half + (0..7)
// elements 8..15 <-> K = 16 + 8*half + (0..7)
__device__ __forceinline__ v16bf ld_frag_a(const bf16* row, int half) {
    v8bf lo = *(const v8bf*)(row + 8 * half);
    v8bf hi = *(const v8bf*)(row + 16 + 8 * half);
    v16bf f;
#pragma unroll
    for (int i = 0; i < 8; ++i) { f[i] = lo[i]; f[i + 8] = hi[i]; }
    return f;
}

// B-fragment (32x16 bf16, KxN), stored N-major in LDS (row = n, K contiguous):
// elements 0..7 <-> K = 8*half + (0..7); elements 8..15 <-> K = 16 + 8*half + (0..7)
__device__ __forceinline__ v16bf ld_frag_b(const bf16* row, int half) {
    v8bf lo = *(const v8bf*)(row + 8 * half);
    v8bf hi = *(const v8bf*)(row + 16 + 8 * half);
    v16bf f;
#pragma unroll
    for (int i = 0; i < 8; ++i) { f[i] = lo[i]; f[i + 8] = hi[i]; }
    return f;
}

// ---------------------------------------------------------------------------
// fp32 -> bf16 convert (same layout)
// ---------------------------------------------------------------------------
__global__ void cvt_f32_bf16(const float* __restrict__ in, bf16* __restrict__ out, long long n) {
    long long i = (long long)blockIdx.x * blockDim.x + threadIdx.x;
    if (i < n) out[i] = (bf16)in[i];
}

// fp32 [K][N] -> bf16 [N][K] (transpose so GEMM B operand is K-contiguous)
__global__ void cvt_transpose(const float* __restrict__ in, bf16* __restrict__ out, int K, int N) {
    int n = blockIdx.x * 32 + (threadIdx.x & 31);
    int k = blockIdx.y * 8 + (threadIdx.x >> 5);
    if (k < K && n < N) out[(size_t)n * K + k] = (bf16)in[(size_t)k * N + n];
}

// ---------------------------------------------------------------------------
// bf16 GEMM: C[M][N] = A[M][K] * Bt[N][K]^T , f32 accumulate.
// Block tile 128x64, BK=32, double-buffered LDS software pipeline.
// 256 threads = 8 waves (4x2); wave tile 32x32 -> 4 independent WMMAs/k-step.
// ---------------------------------------------------------------------------
#define GBK 32
#define LDT 40               // 32 + 8 pad (80B rows -> 16B aligned)
#define ASZ (128 * LDT)
#define BSZ (64 * LDT)

__global__ __launch_bounds__(256) void gemm_bf16_kernel(
    const bf16* __restrict__ A, const bf16* __restrict__ Bt,
    float* __restrict__ Cf, bf16* __restrict__ Cb,
    int M, int N, int K)
{
    __shared__ __align__(16) bf16 As[2 * ASZ];
    __shared__ __align__(16) bf16 Bs[2 * BSZ];

    const int tid  = threadIdx.x;
    const int wave = tid >> 5, lane = tid & 31;
    const int half = lane >> 4, lm = lane & 15;
    const int wm = wave >> 1, wn = wave & 1;        // 4 x 2 waves
    const int m0 = blockIdx.y * 128, n0 = blockIdx.x * 64;

    // cooperative load mapping
    const int arow = tid >> 1;           // 0..127
    const int ak   = (tid & 1) * 16;     // 0 or 16 (two v8bf each)
    const int brow = tid >> 2;           // 0..63
    const int bk   = (tid & 3) * 8;      // 0,8,16,24

    const int gmA = m0 + arow;
    const int gnB = n0 + brow;
    const bool mok = (gmA < M);
    const bool nok = (gnB < N);
    const bf16* aptr = A + (size_t)gmA * K + ak;
    const bf16* bptr = Bt + (size_t)gnB * K + bk;

    v8f acc00 = zero_v8f(), acc01 = zero_v8f();
    v8f acc10 = zero_v8f(), acc11 = zero_v8f();

    v8bf ra0, ra1, rb0;
    auto gload = [&](int k0) {
        ra0 = zero_v8bf(); ra1 = zero_v8bf(); rb0 = zero_v8bf();
        if (mok) {
            ra0 = *(const v8bf*)(aptr + k0);
            ra1 = *(const v8bf*)(aptr + k0 + 8);
        }
        if (nok) rb0 = *(const v8bf*)(bptr + k0);
        if (k0 + GBK < K) {
            if (mok) __builtin_prefetch(aptr + k0 + GBK, 0, 1);
            if (nok) __builtin_prefetch(bptr + k0 + GBK, 0, 1);
        }
    };

    gload(0);

    int buf = 0;
    for (int k0 = 0; k0 < K; k0 += GBK) {
        // stage registers -> LDS[buf]
        *(v8bf*)(As + buf * ASZ + arow * LDT + ak)     = ra0;
        *(v8bf*)(As + buf * ASZ + arow * LDT + ak + 8) = ra1;
        *(v8bf*)(Bs + buf * BSZ + brow * LDT + bk)     = rb0;
        __syncthreads();

        // issue next tile's global loads (latency hidden behind WMMAs)
        if (k0 + GBK < K) gload(k0 + GBK);

        const bf16* abase = As + buf * ASZ + (wm * 32 + lm) * LDT;
        const bf16* bbase = Bs + buf * BSZ + (wn * 32 + lm) * LDT;
        v16bf af0 = ld_frag_a(abase, half);
        v16bf af1 = ld_frag_a(abase + 16 * LDT, half);
        v16bf bf0 = ld_frag_b(bbase, half);
        v16bf bf1 = ld_frag_b(bbase + 16 * LDT, half);
        acc00 = __builtin_amdgcn_wmma_f32_16x16x32_bf16(false, af0, false, bf0, (short)0, acc00, false, false);
        acc01 = __builtin_amdgcn_wmma_f32_16x16x32_bf16(false, af0, false, bf1, (short)0, acc01, false, false);
        acc10 = __builtin_amdgcn_wmma_f32_16x16x32_bf16(false, af1, false, bf0, (short)0, acc10, false, false);
        acc11 = __builtin_amdgcn_wmma_f32_16x16x32_bf16(false, af1, false, bf1, (short)0, acc11, false, false);
        buf ^= 1;
    }

    // Store: D layout -> row m = r + 8*half, col n = lane%16 (+16 second tile)
#pragma unroll
    for (int m16 = 0; m16 < 2; ++m16) {
#pragma unroll
        for (int r = 0; r < 8; ++r) {
            int gm = m0 + wm * 32 + m16 * 16 + half * 8 + r;
            if (gm >= M) continue;
            int gn0 = n0 + wn * 32 + lm;
            int gn1 = gn0 + 16;
            float x0 = (m16 == 0) ? acc00[r] : acc10[r];
            float x1 = (m16 == 0) ? acc01[r] : acc11[r];
            if (Cf) {
                if (gn0 < N) Cf[(size_t)gm * N + gn0] = x0;
                if (gn1 < N) Cf[(size_t)gm * N + gn1] = x1;
            } else {
                if (gn0 < N) Cb[(size_t)gm * N + gn0] = (bf16)x0;
                if (gn1 < N) Cb[(size_t)gm * N + gn1] = (bf16)x1;
            }
        }
    }
}

// ---------------------------------------------------------------------------
// q combine: q[b,s,h,d] = sum_r Aq[b,s,h,r] * rope(Bq[b,s,h,r,:])[d]
// one thread per (b,s,h,d2), d2 in [0,64)
// ---------------------------------------------------------------------------
__global__ void q_combine(const float* __restrict__ Aq, const bf16* __restrict__ Bq,
                          const float* __restrict__ fcos, const float* __restrict__ fsin,
                          bf16* __restrict__ qout)
{
    long long i = (long long)blockIdx.x * blockDim.x + threadIdx.x;
    if (i >= (long long)Bc_ * Sc_ * Hc_ * 64) return;
    int d2 = (int)(i & 63);
    int h  = (int)((i >> 6) & (Hc_ - 1));
    long long bs = i >> 10;                     // b*S + s
    int s = (int)(bs & (Sc_ - 1));
    float c  = fcos[(size_t)s * 64 + d2];
    float sn = fsin[(size_t)s * 64 + d2];
    const float* a  = Aq + bs * (Hc_ * RQc_) + h * RQc_;
    const bf16*  bq = Bq + bs * (Hc_ * RQc_ * Dc_) + (size_t)h * RQc_ * Dc_;
    float q1 = 0.f, q2 = 0.f;
#pragma unroll
    for (int r = 0; r < RQc_; ++r) {
        float x1 = (float)bq[r * Dc_ + d2];
        float x2 = (float)bq[r * Dc_ + 64 + d2];
        q1 += a[r] * (x1 * c - x2 * sn);
        q2 += a[r] * (x1 * sn + x2 * c);
    }
    bf16* qo = qout + ((bs * Hc_ + h) << 7);
    qo[d2] = (bf16)q1;
    qo[64 + d2] = (bf16)q2;
}

// ---------------------------------------------------------------------------
// k/v combine: k[b,s,g,d] = sum_r Ak[b,s,g,r] * rope(Bk[b,s,r,:])[d]; v no rope
// ---------------------------------------------------------------------------
__global__ void kv_combine(const float* __restrict__ Ak, const bf16* __restrict__ Bk,
                           const float* __restrict__ Av, const bf16* __restrict__ Bv,
                           const float* __restrict__ fcos, const float* __restrict__ fsin,
                           bf16* __restrict__ kout, bf16* __restrict__ vout)
{
    long long i = (long long)blockIdx.x * blockDim.x + threadIdx.x;
    if (i >= (long long)Bc_ * Sc_ * HKVc_ * 64) return;
    int d2 = (int)(i & 63);
    int g  = (int)((i >> 6) & (HKVc_ - 1));
    long long bs = i >> 9;
    int s = (int)(bs & (Sc_ - 1));
    float c  = fcos[(size_t)s * 64 + d2];
    float sn = fsin[(size_t)s * 64 + d2];
    const float* ak = Ak + bs * (HKVc_ * RKc_) + g * RKc_;
    const float* av = Av + bs * (HKVc_ * RVc_) + g * RVc_;
    const bf16*  bk = Bk + bs * (RKc_ * Dc_);
    const bf16*  bv = Bv + bs * (RVc_ * Dc_);
    float k1 = 0.f, k2 = 0.f, v1 = 0.f, v2 = 0.f;
#pragma unroll
    for (int r = 0; r < 2; ++r) {
        float x1 = (float)bk[r * Dc_ + d2];
        float x2 = (float)bk[r * Dc_ + 64 + d2];
        k1 += ak[r] * (x1 * c - x2 * sn);
        k2 += ak[r] * (x1 * sn + x2 * c);
        v1 += av[r] * (float)bv[r * Dc_ + d2];
        v2 += av[r] * (float)bv[r * Dc_ + 64 + d2];
    }
    bf16* ko = kout + ((bs * HKVc_ + g) << 7);
    bf16* vo = vout + ((bs * HKVc_ + g) << 7);
    ko[d2] = (bf16)k1; ko[64 + d2] = (bf16)k2;
    vo[d2] = (bf16)v1; vo[64 + d2] = (bf16)v2;
}

// ---------------------------------------------------------------------------
// Flash attention, sliding window + tanh softcap. Br=64 (4 waves x 16 rows),
// Bc=64, D=128. QK^T and PV via v_wmma_f32_16x16x32_bf16.
// grid: (S/64, B*H), block 128.
// ---------------------------------------------------------------------------
#define QLD 136   // 128 + 8 pad
#define VLD 72    // 64 + 8 pad

__global__ __launch_bounds__(128) void attn_kernel(
    const bf16* __restrict__ Q, const bf16* __restrict__ Kk,
    const bf16* __restrict__ V, bf16* __restrict__ Out)
{
    __shared__ __align__(16) bf16 Qs[64 * QLD];
    __shared__ __align__(16) bf16 Ks[64 * QLD];
    __shared__ __align__(16) bf16 Vt[128 * VLD];      // transposed: [d][key]
    __shared__ __align__(16) bf16 Ps[4 * 16 * VLD];   // per-wave P tile [row][key]

    const int tid  = threadIdx.x;
    const int wave = tid >> 5, lane = tid & 31;
    const int half = lane >> 4, lm = lane & 15;
    const int q0 = blockIdx.x * 64;
    const int bh = blockIdx.y;
    const int b = bh >> 4, h = bh & 15;
    const int g = h >> 1;

    const bf16* Qbase = Q  + (((size_t)b * Sc_) * Hc_   + h) * Dc_;
    const bf16* Kbase = Kk + (((size_t)b * Sc_) * HKVc_ + g) * Dc_;
    const bf16* Vbase = V  + (((size_t)b * Sc_) * HKVc_ + g) * Dc_;

    // load Q tile [64][128]
    {
        int row = tid >> 1;
        int dof = (tid & 1) * 64;
#pragma unroll
        for (int j = 0; j < 8; ++j) {
            v8bf v = *(const v8bf*)(Qbase + (size_t)(q0 + row) * (Hc_ * Dc_) + dof + j * 8);
            *(v8bf*)(Qs + row * QLD + dof + j * 8) = v;
        }
    }

    v8f o[8];
#pragma unroll
    for (int sub = 0; sub < 8; ++sub) o[sub] = zero_v8f();
    float mrow[8], lsum[8];
#pragma unroll
    for (int r = 0; r < 8; ++r) { mrow[r] = -1.0e30f; lsum[r] = 0.0f; }

    int kb_lo = q0 - WINDOWc_ + 1;
    if (kb_lo < 0) kb_lo = 0;
    kb_lo &= ~63;
    const int kb_hi = q0 + 64;

    for (int kb = kb_lo; kb < kb_hi; kb += 64) {
        __syncthreads();
        // load K tile [64][128] and V tile transposed [128][64]
        {
            int row = tid >> 1;
            int dof = (tid & 1) * 64;
#pragma unroll
            for (int j = 0; j < 8; ++j) {
                v8bf kv = *(const v8bf*)(Kbase + (size_t)(kb + row) * (HKVc_ * Dc_) + dof + j * 8);
                *(v8bf*)(Ks + row * QLD + dof + j * 8) = kv;
                v8bf vv = *(const v8bf*)(Vbase + (size_t)(kb + row) * (HKVc_ * Dc_) + dof + j * 8);
#pragma unroll
                for (int e = 0; e < 8; ++e)
                    Vt[(dof + j * 8 + e) * VLD + row] = vv[e];
            }
        }
        __syncthreads();

        // S = Q K^T : wave's 16 rows x 64 keys
        v8f sc[4];
#pragma unroll
        for (int sub = 0; sub < 4; ++sub) sc[sub] = zero_v8f();
        const bf16* qrow = Qs + (wave * 16 + lm) * QLD;
#pragma unroll
        for (int kk = 0; kk < 128; kk += 32) {
            v16bf af = ld_frag_a(qrow + kk, half);
#pragma unroll
            for (int sub = 0; sub < 4; ++sub) {
                v16bf bfg = ld_frag_b(Ks + (sub * 16 + lm) * QLD + kk, half);
                sc[sub] = __builtin_amdgcn_wmma_f32_16x16x32_bf16(false, af, false, bfg, (short)0, sc[sub], false, false);
            }
        }

        // softcap + window mask + online softmax (stats per row via intra-half shfl)
#pragma unroll
        for (int r = 0; r < 8; ++r) {
            int qg = q0 + wave * 16 + half * 8 + r;
            float mx = -3.0e38f;
#pragma unroll
            for (int sub = 0; sub < 4; ++sub) {
                int kg = kb + sub * 16 + lm;
                float x = sc[sub][r] * SCALINGc_;
                x = SOFTCAPc_ * tanhf(x * (1.0f / SOFTCAPc_));
                bool ok = (kg <= qg) && ((qg - kg) < WINDOWc_);
                x = ok ? x : -3.0e38f;
                sc[sub][r] = x;
                mx = fmaxf(mx, x);
            }
#pragma unroll
            for (int ofs = 1; ofs < 16; ofs <<= 1)
                mx = fmaxf(mx, __shfl_xor(mx, ofs, 32));
            float mnew  = fmaxf(mrow[r], mx);
            float alpha = __expf(mrow[r] - mnew);
            mrow[r] = mnew;
            float sum = 0.f;
#pragma unroll
            for (int sub = 0; sub < 4; ++sub) {
                float p = __expf(sc[sub][r] - mnew);
                sc[sub][r] = p;
                sum += p;
            }
#pragma unroll
            for (int ofs = 1; ofs < 16; ofs <<= 1)
                sum += __shfl_xor(sum, ofs, 32);
            lsum[r] = lsum[r] * alpha + sum;
#pragma unroll
            for (int sub = 0; sub < 8; ++sub) o[sub][r] *= alpha;
        }

        // P (C-layout) -> LDS -> A-layout fragments for PV
        bf16* pbase = Ps + wave * 16 * VLD;
#pragma unroll
        for (int r = 0; r < 8; ++r) {
            int m = half * 8 + r;
#pragma unroll
            for (int sub = 0; sub < 4; ++sub)
                pbase[m * VLD + sub * 16 + lm] = (bf16)sc[sub][r];
        }
        asm volatile("s_wait_dscnt 0" ::: "memory");

        // O += P * V : 16 rows x 128 D, contraction over 64 keys
#pragma unroll
        for (int kk = 0; kk < 64; kk += 32) {
            v16bf af = ld_frag_a(pbase + lm * VLD + kk, half);
#pragma unroll
            for (int sub = 0; sub < 8; ++sub) {
                v16bf bfg = ld_frag_b(Vt + (sub * 16 + lm) * VLD + kk, half);
                o[sub] = __builtin_amdgcn_wmma_f32_16x16x32_bf16(false, af, false, bfg, (short)0, o[sub], false, false);
            }
        }
    }

    // epilogue: O /= l, store to attn_out[b, s, h*128 + d] (bf16)
    bf16* obase = Out + ((size_t)(b * Sc_ + q0 + wave * 16)) * (Hc_ * Dc_) + h * Dc_;
#pragma unroll
    for (int r = 0; r < 8; ++r) {
        float inv = 1.0f / lsum[r];
        int m = half * 8 + r;
#pragma unroll
        for (int sub = 0; sub < 8; ++sub)
            obase[(size_t)m * (Hc_ * Dc_) + sub * 16 + lm] = (bf16)(o[sub][r] * inv);
    }
}

// ---------------------------------------------------------------------------
// Host-side launch
// ---------------------------------------------------------------------------
extern "C" void kernel_launch(void* const* d_in, const int* in_sizes, int n_in,
                              void* d_out, int out_size, void* d_ws, size_t ws_size,
                              hipStream_t stream)
{
    (void)in_sizes; (void)n_in; (void)out_size; (void)ws_size;
    const float* X    = (const float*)d_in[0];
    const float* WAq  = (const float*)d_in[1];
    const float* WBq  = (const float*)d_in[2];
    const float* WAk  = (const float*)d_in[3];
    const float* WBk  = (const float*)d_in[4];
    const float* WAv  = (const float*)d_in[5];
    const float* WBv  = (const float*)d_in[6];
    const float* Wo   = (const float*)d_in[7];
    const float* fcos = (const float*)d_in[8];
    const float* fsin = (const float*)d_in[9];
    // d_in[10] (mask) and d_in[11] (kv_write_indices) recomputed/implicit.

    const long long MN = (long long)Bc_ * Sc_;   // 4096
    const int K = HIDc_;                         // 2048

    size_t off = 0;
    auto carve = [&](size_t bytes) -> void* {
        off = (off + 255) & ~(size_t)255;
        void* p = (char*)d_ws + off;
        off += bytes;
        return p;
    };

    bf16* X16   = (bf16*)carve((size_t)MN * K * 2);
    bf16* WtAq  = (bf16*)carve((size_t)(Hc_ * RQc_) * K * 2);
    bf16* WtBq  = (bf16*)carve((size_t)(Hc_ * RQc_ * Dc_) * K * 2);
    bf16* WtAk  = (bf16*)carve((size_t)(HKVc_ * RKc_) * K * 2);
    bf16* WtBk  = (bf16*)carve((size_t)(RKc_ * Dc_) * K * 2);
    bf16* WtAv  = (bf16*)carve((size_t)(HKVc_ * RVc_) * K * 2);
    bf16* WtBv  = (bf16*)carve((size_t)(RVc_ * Dc_) * K * 2);
    bf16* WtO   = (bf16*)carve((size_t)HIDc_ * (Hc_ * Dc_) * 2);
    float* Aqf  = (float*)carve((size_t)MN * (Hc_ * RQc_) * 4);
    bf16* Bq16  = (bf16*)carve((size_t)MN * (Hc_ * RQc_ * Dc_) * 2);
    float* Akf  = (float*)carve((size_t)MN * (HKVc_ * RKc_) * 4);
    bf16* Bk16  = (bf16*)carve((size_t)MN * (RKc_ * Dc_) * 2);
    float* Avf  = (float*)carve((size_t)MN * (HKVc_ * RVc_) * 4);
    bf16* Bv16  = (bf16*)carve((size_t)MN * (RVc_ * Dc_) * 2);
    bf16* qb    = (bf16*)carve((size_t)MN * Hc_ * Dc_ * 2);
    bf16* kb    = (bf16*)carve((size_t)MN * HKVc_ * Dc_ * 2);
    bf16* vb    = (bf16*)carve((size_t)MN * HKVc_ * Dc_ * 2);
    bf16* att16 = (bf16*)carve((size_t)MN * Hc_ * Dc_ * 2);

    // 1) convert X to bf16
    {
        long long n = MN * K;
        cvt_f32_bf16<<<(unsigned)((n + 255) / 256), 256, 0, stream>>>(X, X16, n);
    }
    // 2) convert + transpose weights to [N][K] bf16
    auto xpose = [&](const float* w, bf16* wt, int kk, int nn) {
        dim3 g((nn + 31) / 32, (kk + 7) / 8);
        cvt_transpose<<<g, 256, 0, stream>>>(w, wt, kk, nn);
    };
    xpose(WAq, WtAq, K, Hc_ * RQc_);
    xpose(WBq, WtBq, K, Hc_ * RQc_ * Dc_);
    xpose(WAk, WtAk, K, HKVc_ * RKc_);
    xpose(WBk, WtBk, K, RKc_ * Dc_);
    xpose(WAv, WtAv, K, HKVc_ * RVc_);
    xpose(WBv, WtBv, K, RVc_ * Dc_);
    xpose(Wo,  WtO,  Hc_ * Dc_, HIDc_);

    // 3) projection GEMMs (WMMA)
    auto gemm = [&](const bf16* A, const bf16* Bt, float* Cf, bf16* Cb, int M, int N, int KK) {
        dim3 g((N + 63) / 64, (M + 127) / 128);
        gemm_bf16_kernel<<<g, 256, 0, stream>>>(A, Bt, Cf, Cb, M, N, KK);
    };
    gemm(X16, WtAq, Aqf, nullptr, (int)MN, Hc_ * RQc_,        K);
    gemm(X16, WtBq, nullptr, Bq16, (int)MN, Hc_ * RQc_ * Dc_, K);
    gemm(X16, WtAk, Akf, nullptr, (int)MN, HKVc_ * RKc_,      K);
    gemm(X16, WtBk, nullptr, Bk16, (int)MN, RKc_ * Dc_,       K);
    gemm(X16, WtAv, Avf, nullptr, (int)MN, HKVc_ * RVc_,      K);
    gemm(X16, WtBv, nullptr, Bv16, (int)MN, RVc_ * Dc_,       K);

    // 4) RoPE + rank contraction -> q, k, v (bf16)
    {
        long long n = (long long)Bc_ * Sc_ * Hc_ * 64;
        q_combine<<<(unsigned)((n + 255) / 256), 256, 0, stream>>>(Aqf, Bq16, fcos, fsin, qb);
        long long n2 = (long long)Bc_ * Sc_ * HKVc_ * 64;
        kv_combine<<<(unsigned)((n2 + 255) / 256), 256, 0, stream>>>(Akf, Bk16, Avf, Bv16, fcos, fsin, kb, vb);
    }

    // 5) flash attention (WMMA)
    {
        dim3 g(Sc_ / 64, Bc_ * Hc_);
        attn_kernel<<<g, 128, 0, stream>>>(qb, kb, vb, att16);
    }

    // 6) output projection (WMMA) -> d_out (f32)
    gemm(att16, WtO, (float*)d_out, nullptr, (int)MN, HIDc_, Hc_ * Dc_);
}
